// BaoCypherNet_4861902979572
// MI455X (gfx1250) — matrix operand order repaired and checked by reading, not compiled, not run on hardware
//
#include <hip/hip_runtime.h>
#include <hip/hip_bf16.h>
#include <math.h>

// ---------------------------------------------------------------------------
// Types for CDNA5 WMMA (wave32): v_wmma_f32_16x16x32_bf16
// ---------------------------------------------------------------------------
typedef __attribute__((ext_vector_type(16))) __bf16 v16bf;
typedef __attribute__((ext_vector_type(8)))  __bf16 bf16x8;
typedef __attribute__((ext_vector_type(8)))  float  v8f;

#define B_TREES   2048
#define N_NODES   128
#define M_POS     127      // conv output positions (node 0 is the zero pad slot)
#define IDX_LEN   (3 * M_POS)

// K-chunk staged through LDS per iteration (multiple of 32; divides C)
#define KC   64
#define KCP  72            // padded LDS stride (multiple of 8 -> 16B-aligned frags)

// ---------------------------------------------------------------------------
// Layer-1 input prep: trees [B, C=128, N=128] f32 -> node-major bf16 [B, N, C]
// ---------------------------------------------------------------------------
__global__ __launch_bounds__(256)
void transpose_to_bf16(const float* __restrict__ in, __bf16* __restrict__ out)
{
    const int b = blockIdx.x;
    const float* p = in  + (size_t)b * 128 * 128;
    __bf16*      o = out + (size_t)b * 128 * 128;
    for (int e = threadIdx.x; e < 128 * 128; e += 256) {
        const int node = e >> 7;
        const int c    = e & 127;
        o[e] = (__bf16)p[c * 128 + node];   // strided reads stay L2-resident per tree
    }
}

// ---------------------------------------------------------------------------
// Per-tree gathered tree-conv as bf16 WMMA GEMM. One block per tree.
// K ordering: kappa = k*C + c, so src node-major rows gather contiguously.
//   pre[b, o, m+1] = bias[o] + sum_{c,k} W[o,c,k] * src[b, idx[b,3m+k], c]
// Block: (O_TOTAL/16) wave32s; each wave owns one 16-row tile x 8 col tiles.
// ---------------------------------------------------------------------------
template <int O_TOTAL, int C_IN_T>
__global__ __launch_bounds__((O_TOTAL / 16) * 32)
void tree_conv_wmma(const __bf16* __restrict__ src,  // [B, 128, C_IN_T] node-major
                    const int*   __restrict__ idx,   // [B, 381]
                    const float* __restrict__ W,     // [O_TOTAL, C_IN_T, 3]
                    const float* __restrict__ bias,  // [O_TOTAL]
                    float* __restrict__ pre)         // [B, O_TOTAL, 128] (nodes 1..127)
{
    constexpr int K    = 3 * C_IN_T;
    constexpr int NTHR = (O_TOTAL / 16) * 32;

    __shared__ __bf16 Bs[128 * KCP];       // gathered B chunk, [m][kappa_local]
    __shared__ __bf16 As[O_TOTAL * KCP];   // weight tile, [o][kappa_local]

    const int b    = blockIdx.x;
    const int tid  = threadIdx.x;
    const int lane = tid & 31;
    const int wave = tid >> 5;             // row tile index (16 rows each)

    const int*    tidx = idx + (size_t)b * IDX_LEN;
    const __bf16* tsrc = src + (size_t)b * N_NODES * C_IN_T;

    __builtin_prefetch(tidx, 0, 0);                   // -> global_prefetch_b8
    __builtin_prefetch(tidx + IDX_LEN - 1, 0, 0);

    v8f acc[8];
#pragma unroll
    for (int i = 0; i < 8; ++i)
#pragma unroll
        for (int j = 0; j < 8; ++j) acc[i][j] = 0.0f;

    const int mrow = lane & 15;            // row within A tile / col within B tile
    const int g    = lane >> 4;            // lane half

    for (int k0 = 0; k0 < K; k0 += KC) {
        const int kblk = k0 / C_IN_T;      // kernel tap (chunk lies in one tap)
        const int c0   = k0 - kblk * C_IN_T;
        __syncthreads();

        // ---- gather B chunk: contiguous 8-element row slices per task ----
        for (int t = tid; t < 128 * 8; t += NTHR) {
            const int m  = t >> 3;
            const int cv = (t & 7) * 8;
            bf16x8 v;
#pragma unroll
            for (int i = 0; i < 8; ++i) v[i] = (__bf16)0.0f;
            if (m < M_POS) {
                const int node = tidx[3 * m + kblk];
                v = *(const bf16x8*)(tsrc + node * C_IN_T + c0 + cv);
            }
            *(bf16x8*)(Bs + m * KCP + cv) = v;
        }
        // ---- stage weight tile: As[o][kl] = W[o, c0+kl, kblk] ----
        for (int e = tid; e < O_TOTAL * KC; e += NTHR) {
            const int r  = e >> 6;         // KC == 64
            const int kl = e & 63;
            As[r * KCP + kl] = (__bf16)W[(size_t)r * K + (c0 + kl) * 3 + kblk];
        }
        __syncthreads();

#pragma unroll
        for (int s = 0; s < KC; s += 32) {
            // A fragment (ISA 16-bit A 16x32 layout): lane(m,g)
            //   e<8 : K = s + g*8 + e ; e>=8 : K = s + 16 + g*8 + (e-8)
            const __bf16* arow = &As[(wave * 16 + mrow) * KCP + s];
            bf16x8 alo = *(const bf16x8*)(arow + g * 8);
            bf16x8 ahi = *(const bf16x8*)(arow + 16 + g * 8);
            v16bf afrag;
#pragma unroll
            for (int i = 0; i < 8; ++i) { afrag[i] = alo[i]; afrag[i + 8] = ahi[i]; }

#pragma unroll
            for (int ct = 0; ct < 8; ++ct) {
                // B fragment (ISA 16-bit B 32x16): lane(n,g)
                //   e<8 : K = s + g*8 + e ; e>=8 : K = s + 16 + g*8 + (e-8)
                const __bf16* bcol = &Bs[(ct * 16 + mrow) * KCP + s + g * 8];
                bf16x8 blo = *(const bf16x8*)bcol;
                bf16x8 bhi = *(const bf16x8*)(bcol + 16);
                v16bf bfrag;
#pragma unroll
                for (int i = 0; i < 8; ++i) { bfrag[i] = blo[i]; bfrag[i + 8] = bhi[i]; }

                acc[ct] = __builtin_amdgcn_wmma_f32_16x16x32_bf16(
                    false, afrag, false, bfrag, (short)0, acc[ct], false, false);
            }
        }
    }

    // ---- epilogue: D layout -> lane(n,g): col n, rows r + 8*g; node = col+1 ----
    const int n  = lane & 15;
    const int gl = lane >> 4;
#pragma unroll
    for (int ct = 0; ct < 8; ++ct) {
        const int node = ct * 16 + n + 1;
        if (node < N_NODES) {
#pragma unroll
            for (int r = 0; r < 8; ++r) {
                const int orow = wave * 16 + r + 8 * gl;
                pre[((size_t)b * O_TOTAL + orow) * N_NODES + node] =
                    acc[ct][r] + bias[orow];
            }
        }
    }
}

// ---------------------------------------------------------------------------
// Per-tree layer norm (mean/unbiased-std over O*128, zero slot included as 0)
// + leaky relu, emitting node-major bf16 [128, O_T] for the next gather.
// ---------------------------------------------------------------------------
template <int O_T>
__global__ __launch_bounds__(256)
void tree_norm(const float* __restrict__ pre, __bf16* __restrict__ out)
{
    const int b   = blockIdx.x;
    const int tid = threadIdx.x;
    const float* p = pre + (size_t)b * O_T * N_NODES;
    __bf16*      o = out + (size_t)b * N_NODES * O_T;

    float s = 0.0f, ss = 0.0f;
    for (int e = tid; e < O_T * M_POS; e += 256) {
        const int row  = e / M_POS;
        const int node = e - row * M_POS + 1;
        const float v = p[row * N_NODES + node];
        s += v; ss += v * v;
    }
    __shared__ float rs[256], rss[256];
    rs[tid] = s; rss[tid] = ss;
    __syncthreads();
    for (int d = 128; d > 0; d >>= 1) {
        if (tid < d) { rs[tid] += rs[tid + d]; rss[tid] += rss[tid + d]; }
        __syncthreads();
    }
    const float cnt  = (float)(O_T * N_NODES);
    const float mean = rs[0] / cnt;
    float var = (rss[0] - rs[0] * rs[0] / cnt) / (cnt - 1.0f);
    var = var > 0.0f ? var : 0.0f;
    const float inv = 1.0f / (sqrtf(var) + 1e-5f);

    // node-major write: out[node][c] = leaky((pre[c][node] - mean) * inv)
    for (int e = tid; e < N_NODES * O_T; e += 256) {
        const int node = e / O_T;          // O_T is a power of two
        const int c    = e - node * O_T;
        const float v  = (node == 0) ? 0.0f : p[c * N_NODES + node];
        float y = (v - mean) * inv;
        y = y > 0.0f ? y : 0.01f * y;      // layers 1 & 2 both apply leaky
        o[e] = (__bf16)y;
    }
}

// ---------------------------------------------------------------------------
// Layer-3 norm + max-pool + FC head, fused per tree. Norm is monotone affine,
// so pool the raw pre-activations (seed 0 for the zero slot) then normalize.
// ---------------------------------------------------------------------------
__global__ __launch_bounds__(64)
void pool_head(const float* __restrict__ pre3,          // [B, 64, 128]
               const float* __restrict__ W4, const float* __restrict__ b4,
               const float* __restrict__ W5, const float* __restrict__ b5,
               float* __restrict__ out)                  // [B, 1]
{
    const int b   = blockIdx.x;
    const int tid = threadIdx.x;                         // 0..63 = channel
    const float* row = pre3 + ((size_t)b * 64 + tid) * N_NODES;

    __shared__ float ssum[64], ssq[64], pooled[64], h[32];

    float mx = 0.0f, s = 0.0f, ss = 0.0f;                // node 0 value is 0
    for (int node = 1; node < N_NODES; ++node) {
        const float v = row[node];
        mx = fmaxf(mx, v); s += v; ss += v * v;
    }
    ssum[tid] = s; ssq[tid] = ss;
    __syncthreads();
    for (int d = 32; d > 0; d >>= 1) {
        if (tid < d) { ssum[tid] += ssum[tid + d]; ssq[tid] += ssq[tid + d]; }
        __syncthreads();
    }
    const float cnt  = 64.0f * 128.0f;
    const float mean = ssum[0] / cnt;
    float var = (ssq[0] - ssum[0] * ssum[0] / cnt) / (cnt - 1.0f);
    var = var > 0.0f ? var : 0.0f;
    const float inv = 1.0f / (sqrtf(var) + 1e-5f);

    pooled[tid] = (mx - mean) * inv;
    __syncthreads();

    if (tid < 32) {
        float a = b4[tid];
        for (int i = 0; i < 64; ++i) a += W4[tid * 64 + i] * pooled[i];
        h[tid] = a > 0.0f ? a : 0.01f * a;
    }
    __syncthreads();
    if (tid == 0) {
        float a = b5[0];
        for (int j = 0; j < 32; ++j) a += W5[j] * h[j];
        out[b] = a;
    }
}

// ---------------------------------------------------------------------------
extern "C" void kernel_launch(void* const* d_in, const int* in_sizes, int n_in,
                              void* d_out, int out_size, void* d_ws, size_t ws_size,
                              hipStream_t stream)
{
    const float* trees = (const float*)d_in[0];   // [2048,128,128] f32
    const int*   idx   = (const int*)  d_in[1];   // [2048,381] i32
    const float* W1 = (const float*)d_in[2];  const float* b1 = (const float*)d_in[3];
    const float* W2 = (const float*)d_in[4];  const float* b2 = (const float*)d_in[5];
    const float* W3 = (const float*)d_in[6];  const float* b3 = (const float*)d_in[7];
    const float* W4 = (const float*)d_in[8];  const float* b4 = (const float*)d_in[9];
    const float* W5 = (const float*)d_in[10]; const float* b5 = (const float*)d_in[11];
    float* out = (float*)d_out;

    // Workspace layout (stream-ordered reuse keeps this safe):
    //   [0, 256MB)   : pre-activation f32 (reused by layers 1/2/3)
    //   [256, 384MB) : node-major bf16 activations (layer1 out, then layer2 out)
    //   [384, 448MB) : node-major bf16 transposed input trees
    const size_t PRE_BYTES  = (size_t)B_TREES * 256 * N_NODES * sizeof(float);
    const size_t NORM_BYTES = (size_t)B_TREES * N_NODES * 256 * sizeof(__bf16);
    float*  pre   = (float*)d_ws;
    __bf16* xnorm = (__bf16*)((char*)d_ws + PRE_BYTES);
    __bf16* xin0  = (__bf16*)((char*)d_ws + PRE_BYTES + NORM_BYTES);

    transpose_to_bf16<<<B_TREES, 256, 0, stream>>>(trees, xin0);

    // Layer 1: conv(128 -> 256) + norm + leaky   (16 waves/block)
    tree_conv_wmma<256, 128><<<B_TREES, 512, 0, stream>>>(xin0, idx, W1, b1, pre);
    tree_norm<256><<<B_TREES, 256, 0, stream>>>(pre, xnorm);

    // Layer 2: conv(256 -> 128) + norm + leaky   (8 waves/block)
    tree_conv_wmma<128, 256><<<B_TREES, 256, 0, stream>>>(xnorm, idx, W2, b2, pre);
    tree_norm<128><<<B_TREES, 256, 0, stream>>>(pre, xnorm);

    // Layer 3: conv(128 -> 64) + fused norm/max-pool/head   (4 waves/block)
    tree_conv_wmma<64, 128><<<B_TREES, 128, 0, stream>>>(xnorm, idx, W3, b3, pre);
    pool_head<<<B_TREES, 64, 0, stream>>>(pre, W4, b4, W5, b5, out);
}
